// Model_16673063043377
// MI455X (gfx1250) — matrix operand order repaired and checked by reading, not compiled, not run on hardware
//
#include <hip/hip_runtime.h>

// ---------------------------------------------------------------------------
// Types
// ---------------------------------------------------------------------------
typedef __bf16 bf16_t;
typedef __attribute__((ext_vector_type(16))) __bf16    v16bf;
typedef __attribute__((ext_vector_type(8)))  float     v8f;
typedef __attribute__((ext_vector_type(4)))  float     f32x4;
typedef __attribute__((ext_vector_type(4)))  unsigned  u32x4;
typedef __attribute__((ext_vector_type(8)))  int       i32x8;
typedef __attribute__((ext_vector_type(4)))  int       i32x4;

#define BB      4
#define LSEQ    4096
#define DMODEL  512
#define NH      8
#define NP      4
#define HD      64
#define MROWS   (BB * LSEQ)      // 16384

// LDS W-panel: 64 rows x 512 K, padded 8 elems (16B) per row -> 520 elems/row.
// Row stride = 1040B = 260 banks ; per-lane stride 4 banks -> conflict-free b128.
#define WROW    520

// ---------------------------------------------------------------------------
// fp32 -> bf16 helpers (round-to-nearest-even)
// ---------------------------------------------------------------------------
__device__ __forceinline__ unsigned bf16bits(float x) {
  unsigned u = __builtin_bit_cast(unsigned, x);
  u = (u + 0x7FFFu + ((u >> 16) & 1u)) >> 16;
  return u & 0xFFFFu;
}
__device__ __forceinline__ unsigned bfpack2(float lo, float hi) {
  return bf16bits(lo) | (bf16bits(hi) << 16);
}

// 8 fp32 -> 8 bf16 per thread, 16B store
__global__ __launch_bounds__(256) void cvt8_kernel(const float* __restrict__ s,
                                                   unsigned* __restrict__ d, int n8) {
  int i = blockIdx.x * 256 + threadIdx.x;
  if (i >= n8) return;
  const f32x4* sp = (const f32x4*)s + (size_t)i * 2;
  f32x4 a = sp[0], b = sp[1];
  u32x4 r;
  r[0] = bfpack2(a[0], a[1]);
  r[1] = bfpack2(a[2], a[3]);
  r[2] = bfpack2(b[0], b[1]);
  r[3] = bfpack2(b[2], b[3]);
  ((u32x4*)d)[i] = r;
}

__global__ __launch_bounds__(256) void copyf_kernel(const float* __restrict__ s,
                                                    float* __restrict__ d, int n) {
  int i = blockIdx.x * 256 + threadIdx.x;
  if (i < n) d[i] = s[i];
}

// ---------------------------------------------------------------------------
// WMMA GEMM:  Y[M,N] = X[M,K](bf16) * W[N,K](bf16)^T + bias[N]   (fp32 out)
// Block: 256 threads = 8 waves. Tile: 128 rows x 64 cols, K = 512 fixed shape
// multiples. The 64x512 W panel is DMA'd into LDS ONCE per block by the
// Tensor Data Mover (wave 0 issues, pad_enable gives the 16B row padding),
// then the K-loop runs barrier-free out of LDS.
// Requires: M%128==0, N%64==0, K==512.
// ---------------------------------------------------------------------------
union AFrag { f32x4 f[2]; v16bf v; };

__global__ __launch_bounds__(256) void gemm_bf16_wmma(
    const bf16_t* __restrict__ X, const bf16_t* __restrict__ W,
    const float* __restrict__ bias, float* __restrict__ Y,
    int M, int N, int K)
{
  __shared__ alignas(128) bf16_t wpanel[64 * WROW];   // 66560 B

  const int tid   = threadIdx.x;
  const int wv    = tid >> 5;
  const int lane  = tid & 31;
  const int lmod  = lane & 15;
  const int lhalf = lane >> 4;

  const int m0 = blockIdx.x * 128 + wv * 16;
  const int n0 = blockIdx.y * 64;

  // ---- TDM: stage W[n0..n0+63][0..511] (contiguous 64KB) into LDS, with
  //      16B padding inserted after every 1024B row. One op per block. ----
  if (tid < 32) {
    unsigned long long ga = (unsigned long long)(uintptr_t)(W + (size_t)n0 * 512);
    unsigned ldsa = (unsigned)(uintptr_t)(&wpanel[0]);   // low 32 bits = LDS offset
    u32x4 g0;
    g0[0] = 1u;                                          // count=1 (valid D#)
    g0[1] = ldsa;                                        // lds_addr
    g0[2] = (unsigned)(ga & 0xFFFFFFFFu);                // global_addr[31:0]
    g0[3] = (unsigned)((ga >> 32) & 0x1FFFFFFu)          // global_addr[56:32]
          | (2u << 30);                                  // type = 2 ("image")
    i32x8 g1;
    g1[0] = (int)(0x00010000u                            // data_size = 2B
          | (1u << 20)                                   // pad_enable
          | (7u << 22)                                   // pad_interval: 256 DW (1024B)
          | (3u << 25));                                 // pad_amount: 4 DW (16B)
    g1[1] = (int)(0x8000u << 16);                        // tensor_dim0 = 32768 (lo16)
    g1[2] = (int)(1u << 16);                             // tensor_dim0 hi=0 | tensor_dim1=1 lo
    g1[3] = (int)(0x8000u << 16);                        // tensor_dim1 hi=0 | tile_dim0 = 32768
    g1[4] = 1;                                           // tile_dim1 = 1, tile_dim2 = 0
    g1[5] = 32768;                                       // tensor_dim0_stride lo32
    g1[6] = 0;                                           // stride hi | dim1_stride lo
    g1[7] = 0;
    i32x4 gz4 = {0, 0, 0, 0};
    i32x8 gz8 = {0, 0, 0, 0, 0, 0, 0, 0};
    __builtin_amdgcn_tensor_load_to_lds(g0, g1, gz4, gz4, gz8, 0);
    __builtin_amdgcn_s_wait_tensorcnt(0);
  }
  __syncthreads();   // publish panel to all 8 waves

  const long arowoff = (long)(m0 + lmod) * K;   // A row for this lane
  v8f acc[4] = {v8f{}, v8f{}, v8f{}, v8f{}};

  for (int kk = 0; kk < K; kk += 32) {
    // A fragment: lane<16 -> row m0+lane,  K {kk..kk+7, kk+16..kk+23}
    //             lane>=16-> row m0+lane-16, K {kk+8..15, kk+24..31}
    AFrag a;
    const bf16_t* ap = X + arowoff + kk + lhalf * 8;
    a.f[0] = *(const f32x4*)(ap);
    a.f[1] = *(const f32x4*)(ap + 16);
    __builtin_prefetch((const void*)(ap + 32), 0, 1);   // next K-slab of A

    // Load ALL four B fragments first (distinct regs) so the ds_loads can
    // overlap and the 4 WMMAs issue back-to-back.
    // B fragment: lane holds column n0+t*16+lmod, same K split as A
    //   lanes 0-15: K {kk..kk+7, kk+16..kk+23} ; lanes 16-31: K {kk+8..15, kk+24..31}
    AFrag bfr[4];
    #pragma unroll
    for (int t = 0; t < 4; ++t) {
      const bf16_t* bp = &wpanel[(t * 16 + lmod) * WROW + kk + lhalf * 8];
      bfr[t].f[0] = *(const f32x4*)(bp);
      bfr[t].f[1] = *(const f32x4*)(bp + 16);
    }
    #pragma unroll
    for (int t = 0; t < 4; ++t) {
      acc[t] = __builtin_amdgcn_wmma_f32_16x16x32_bf16(
          false, a.v, false, bfr[t].v, (short)0, acc[t], false, false);
    }
  }

  // C/D layout: VGPR j -> M = m0 + j + (lane>=16 ? 8 : 0), N = n0 + t*16 + lmod
  const int rbase = m0 + (lhalf << 3);
  #pragma unroll
  for (int t = 0; t < 4; ++t) {
    const int   col = n0 + t * 16 + lmod;
    const float bsv = bias[col];
    #pragma unroll
    for (int j = 0; j < 8; ++j) {
      Y[(long)(rbase + j) * N + col] = acc[t][j] + bsv;
    }
  }
}

// ---------------------------------------------------------------------------
// Deformable sampling + softmax(P=4) + weighted V sum.
// One wave32 per (b, h, l); each lane owns dims d=lane and d=lane+32 (D=64).
// Y1 row layout (ld=576): [Q(512: h*64+d) | OFF(32: h*4+p) | AW(32: h*4+p)]
// Y2 row layout (ld=1024): [K(512) | V(512)], rows indexed by b*L + lk.
// Output: bf16 (row, h*64+d) -> feeds the final WMMA GEMM.
// ---------------------------------------------------------------------------
__global__ __launch_bounds__(256) void attn_sample_kernel(
    const float* __restrict__ Y1, const float* __restrict__ Y2,
    unsigned short* __restrict__ Aout)
{
  const int lane  = threadIdx.x & 31;
  const int gwave = blockIdx.x * 8 + (threadIdx.x >> 5);
  const int h     = gwave & (NH - 1);
  const int row   = gwave >> 3;            // b*L + l
  const int l     = row & (LSEQ - 1);
  const int bb    = row >> 12;             // / 4096

  const long y1row = (long)row * 576;
  const float q0 = Y1[y1row + h * HD + lane];
  const float q1 = Y1[y1row + h * HD + lane + 32];

  int   i0s[NP], i1s[NP];
  float w0s[NP], w1s[NP], logit[NP];

  #pragma unroll
  for (int p = 0; p < NP; ++p) {
    const float off = Y1[y1row + 512 + h * NP + p];
    const float aw  = Y1[y1row + 544 + h * NP + p];
    float idx = fminf(fmaxf((float)l + off, 0.0f), (float)(LSEQ - 1));
    int i0 = (int)idx;                       // floor (idx >= 0)
    if (i0 > LSEQ - 1) i0 = LSEQ - 1;
    int i1 = i0 + 1; if (i1 > LSEQ - 1) i1 = LSEQ - 1;
    const float w1 = idx - (float)i0;
    const float w0 = 1.0f - w1;

    const long r0 = ((long)(bb * LSEQ + i0)) * 1024 + h * HD;
    const long r1 = ((long)(bb * LSEQ + i1)) * 1024 + h * HD;
    const float k0 = w0 * Y2[r0 + lane]      + w1 * Y2[r1 + lane];
    const float k1 = w0 * Y2[r0 + lane + 32] + w1 * Y2[r1 + lane + 32];

    float part = q0 * k0 + q1 * k1;          // lane-partial dot over 2 dims
    #pragma unroll
    for (int s = 16; s > 0; s >>= 1) part += __shfl_xor(part, s, 32);

    i0s[p] = i0; i1s[p] = i1; w0s[p] = w0; w1s[p] = w1;
    logit[p] = part * 0.125f + aw;           // 1/sqrt(64) = 0.125
  }

  // softmax over P=4 (replicated in all lanes)
  const float m  = fmaxf(fmaxf(logit[0], logit[1]), fmaxf(logit[2], logit[3]));
  const float e0 = __expf(logit[0] - m), e1 = __expf(logit[1] - m);
  const float e2 = __expf(logit[2] - m), e3 = __expf(logit[3] - m);
  const float inv = 1.0f / (e0 + e1 + e2 + e3);
  const float wt[NP] = {e0 * inv, e1 * inv, e2 * inv, e3 * inv};

  float o0 = 0.0f, o1 = 0.0f;
  #pragma unroll
  for (int p = 0; p < NP; ++p) {
    const long r0 = ((long)(bb * LSEQ + i0s[p])) * 1024 + 512 + h * HD;
    const long r1 = ((long)(bb * LSEQ + i1s[p])) * 1024 + 512 + h * HD;
    o0 += wt[p] * (w0s[p] * Y2[r0 + lane]      + w1s[p] * Y2[r1 + lane]);
    o1 += wt[p] * (w0s[p] * Y2[r0 + lane + 32] + w1s[p] * Y2[r1 + lane + 32]);
  }

  const long obase = (long)row * DMODEL + h * HD + lane;
  Aout[obase]      = (unsigned short)bf16bits(o0);
  Aout[obase + 32] = (unsigned short)bf16bits(o1);
}

// ---------------------------------------------------------------------------
// Host-side orchestration
// ---------------------------------------------------------------------------
extern "C" void kernel_launch(void* const* d_in, const int* in_sizes, int n_in,
                              void* d_out, int out_size, void* d_ws, size_t ws_size,
                              hipStream_t stream) {
  (void)in_sizes; (void)n_in; (void)out_size; (void)ws_size;

  const float* q_in  = (const float*)d_in[0];
  const float* kv_in = (const float*)d_in[1];
  const float* Wq    = (const float*)d_in[2];
  const float* bq    = (const float*)d_in[3];
  const float* Wk    = (const float*)d_in[4];
  const float* bk    = (const float*)d_in[5];
  const float* Wv    = (const float*)d_in[6];
  const float* bv    = (const float*)d_in[7];
  const float* Woff  = (const float*)d_in[8];
  const float* boff  = (const float*)d_in[9];
  const float* Waw   = (const float*)d_in[10];
  const float* baw   = (const float*)d_in[11];
  const float* Wout  = (const float*)d_in[12];
  const float* bout  = (const float*)d_in[13];
  float* out = (float*)d_out;

  // workspace carve-out (256B aligned)
  char* ws = (char*)d_ws;
  size_t o = 0;
  auto take = [&](size_t bytes) -> char* {
    char* p = ws + o;
    o += (bytes + 255) & ~(size_t)255;
    return p;
  };
  bf16_t* Xq  = (bf16_t*)take((size_t)MROWS * DMODEL * 2);   // q_in  bf16
  bf16_t* Xkv = (bf16_t*)take((size_t)MROWS * DMODEL * 2);   // kv_in bf16
  bf16_t* W1  = (bf16_t*)take((size_t)576  * DMODEL * 2);    // [Wq;Woff;Waw]
  bf16_t* W2  = (bf16_t*)take((size_t)1024 * DMODEL * 2);    // [Wk;Wv]
  bf16_t* W3  = (bf16_t*)take((size_t)DMODEL * DMODEL * 2);  // Wout
  float*  b1  = (float*)take((size_t)576 * 4);               // [bq;boff;baw]
  float*  b2  = (float*)take((size_t)1024 * 4);              // [bk;bv]
  float*  Y1  = (float*)take((size_t)MROWS * 576 * 4);       // Q|OFF|AW
  float*  Y2  = (float*)take((size_t)MROWS * 1024 * 4);      // K|V
  unsigned short* Att = (unsigned short*)take((size_t)MROWS * DMODEL * 2);

  // --- 1. fp32 -> bf16 conversions / weight packing ---
  {
    int n8 = MROWS * DMODEL / 8;   // 1,048,576
    cvt8_kernel<<<n8 / 256, 256, 0, stream>>>(q_in,  (unsigned*)Xq,  n8);
    cvt8_kernel<<<n8 / 256, 256, 0, stream>>>(kv_in, (unsigned*)Xkv, n8);
  }
  {
    int n8w = DMODEL * DMODEL / 8;   // 32768
    int n8s = 32 * DMODEL / 8;       // 2048
    cvt8_kernel<<<n8w / 256, 256, 0, stream>>>(Wq,   (unsigned*)(W1),                n8w);
    cvt8_kernel<<<n8s / 256, 256, 0, stream>>>(Woff, (unsigned*)(W1 + 512 * DMODEL), n8s);
    cvt8_kernel<<<n8s / 256, 256, 0, stream>>>(Waw,  (unsigned*)(W1 + 544 * DMODEL), n8s);
    cvt8_kernel<<<n8w / 256, 256, 0, stream>>>(Wk,   (unsigned*)(W2),                n8w);
    cvt8_kernel<<<n8w / 256, 256, 0, stream>>>(Wv,   (unsigned*)(W2 + 512 * DMODEL), n8w);
    cvt8_kernel<<<n8w / 256, 256, 0, stream>>>(Wout, (unsigned*)(W3),                n8w);
  }
  copyf_kernel<<<3, 256, 0, stream>>>(bq,   b1,       512);
  copyf_kernel<<<1, 256, 0, stream>>>(boff, b1 + 512, 32);
  copyf_kernel<<<1, 256, 0, stream>>>(baw,  b1 + 544, 32);
  copyf_kernel<<<3, 256, 0, stream>>>(bk,   b2,       512);
  copyf_kernel<<<3, 256, 0, stream>>>(bv,   b2 + 512, 512);

  // --- 2. projection GEMMs (WMMA bf16 -> fp32) ---
  gemm_bf16_wmma<<<dim3(MROWS / 128, 576 / 64), 256, 0, stream>>>(
      Xq, W1, b1, Y1, MROWS, 576, DMODEL);
  gemm_bf16_wmma<<<dim3(MROWS / 128, 1024 / 64), 256, 0, stream>>>(
      Xkv, W2, b2, Y2, MROWS, 1024, DMODEL);

  // --- 3. deformable sampling + softmax + V aggregation ---
  attn_sample_kernel<<<(MROWS * NH) / 8, 256, 0, stream>>>(Y1, Y2, Att);

  // --- 4. output projection GEMM -> d_out ---
  gemm_bf16_wmma<<<dim3(MROWS / 128, DMODEL / 64), 256, 0, stream>>>(
      (const bf16_t*)Att, W3, bout, out, MROWS, DMODEL, DMODEL);
}
